// RSSM_90769838833917
// MI455X (gfx1250) — matrix operand order repaired and checked by reading, not compiled, not run on hardware
//
#include <hip/hip_runtime.h>
#include <stdint.h>

typedef __bf16 bf16_t;
typedef bf16_t v16bf __attribute__((ext_vector_type(16)));
typedef bf16_t v8bf  __attribute__((ext_vector_type(8)));
typedef float  v8f   __attribute__((ext_vector_type(8)));

#define B_    1024
#define T_    64
#define ENC_  1024
#define ACT_  16
#define S_    30
#define D_    32
#define SD_   960
#define DET_  200
#define HID_  200
#define ENS_  5

// padded dims (K multiples of 32, N multiples of 16)
#define HP_    208   // HID/DET padded
#define KIMG_  992   // 976 -> 992
#define KGRU_  416   // 208 + 208
#define NGRU_  624   // 3 * 208
#define KENS_  224   // 208 -> 224
#define KOBS_  1248  // 208 + 1024 + 16 pad
#define NT13   13
#define NT39   39
#define NT60   60

// ---------------- helpers ----------------

__device__ inline bf16_t f2bf(float f) {
  uint32_t u = __float_as_uint(f);
  uint32_t r = (u + 0x7FFFu + ((u >> 16) & 1u)) >> 16;  // RNE
  return __builtin_bit_cast(bf16_t, (uint16_t)r);
}

__device__ inline v8f zero8() {
  v8f z;
#pragma unroll
  for (int j = 0; j < 8; ++j) z[j] = 0.f;
  return z;
}

__device__ inline v8f wmma_bf16(v16bf a, v16bf b, v8f c) {
  return __builtin_amdgcn_wmma_f32_16x16x32_bf16(false, a, false, b, (short)0, c,
                                                 false, false);
}

// A fragment for v_wmma_*_16x16x32 (16-bit): lane<16 -> K +0..7 / +16..23,
// lane>=16 -> K +8..15 / +24..31, row = lane&15 (rowbase points at that row).
__device__ inline v16bf load_afrag(const bf16_t* rowbase, int kb, int lane) {
  int off = kb + ((lane & 16) ? 8 : 0);
  v8bf lo = *(const v8bf*)(rowbase + off);
  v8bf hi = *(const v8bf*)(rowbase + off + 16);
  v16bf a;
#pragma unroll
  for (int j = 0; j < 8; ++j) { a[j] = lo[j]; a[8 + j] = hi[j]; }
  return a;
}

// ---------------- JAX threefry2x32 ----------------

struct U2 { uint32_t x, y; };

__device__ inline uint32_t rotl32(uint32_t v, int r) { return (v << r) | (v >> (32 - r)); }

__device__ inline U2 tf2x32(uint32_t k0, uint32_t k1, uint32_t x0, uint32_t x1) {
  uint32_t k2 = k0 ^ k1 ^ 0x1BD11BDAu;
  const int RA[4] = {13, 15, 26, 6};
  const int RB[4] = {17, 29, 16, 24};
  x0 += k0; x1 += k1;
#pragma unroll
  for (int i = 0; i < 4; ++i) { x0 += x1; x1 = rotl32(x1, RA[i]); x1 ^= x0; }
  x0 += k1; x1 += k2 + 1u;
#pragma unroll
  for (int i = 0; i < 4; ++i) { x0 += x1; x1 = rotl32(x1, RB[i]); x1 ^= x0; }
  x0 += k2; x1 += k0 + 2u;
#pragma unroll
  for (int i = 0; i < 4; ++i) { x0 += x1; x1 = rotl32(x1, RA[i]); x1 ^= x0; }
  x0 += k0; x1 += k1 + 3u;
#pragma unroll
  for (int i = 0; i < 4; ++i) { x0 += x1; x1 = rotl32(x1, RB[i]); x1 ^= x0; }
  x0 += k1; x1 += k2 + 4u;
#pragma unroll
  for (int i = 0; i < 4; ++i) { x0 += x1; x1 = rotl32(x1, RA[i]); x1 ^= x0; }
  x0 += k2; x1 += k0 + 5u;
  U2 r; r.x = x0; r.y = x1; return r;
}

__device__ inline float gumbel_elem(uint32_t kp0, uint32_t kp1, uint32_t e) {
  const uint32_t h = (uint32_t)(B_ * S_ * D_) / 2u;  // 491520
  uint32_t bits;
  if (e < h) bits = tf2x32(kp0, kp1, e, e + h).x;
  else       bits = tf2x32(kp0, kp1, e - h, e).y;
  float f = __uint_as_float(0x3F800000u | (bits >> 9)) - 1.0f;  // [0,1)
  const float tiny = 1.17549435e-38f;
  float u = f * (1.0f - tiny) + tiny;
  return -__logf(-__logf(u));
}

// ---------------- weight packing ----------------
// packed bf16 fragment layout: idx = ((kc*NT + nt)*32 + lane)*16 + j
// value = W[kc*32 + ((lane&16)?8:0) + (j&7) + ((j&8)?16:0)][nt*16 + (lane&15)]  (K rows, N cols)

__global__ void pack_w(int mode, const float* __restrict__ src,
                       bf16_t* __restrict__ dst, int total) {
  int p = blockIdx.x * 256 + threadIdx.x;
  int e = blockIdx.y;
  if (p >= total) return;
  int j = p & 15;
  int lane = (p >> 4) & 31;
  int rest = p >> 9;
  int NT = (mode == 1) ? NT39 : ((mode == 3 || mode == 5) ? NT60 : NT13);
  int nt = rest % NT;
  int kc = rest / NT;
  int k = kc * 32 + ((lane & 16) ? 8 : 0) + (j & 7) + ((j & 8) ? 16 : 0);
  int n = nt * 16 + (lane & 15);
  float v = 0.f;
  switch (mode) {
    case 0:  // W_img [976,200] -> [992,208]
      if (k < 976 && n < 200) v = src[(size_t)k * 200 + n];
      break;
    case 1: {  // W_gru [400,600] -> [416,624], rows/cols remapped per gate
      int g = n / HP_, c = n - g * HP_;
      int kr = (k < 200) ? k : ((k >= HP_ && k < HP_ + 200) ? (k - 8) : -1);
      if (kr >= 0 && c < 200) v = src[(size_t)kr * 600 + g * 200 + c];
      break; }
    case 2:  // W_out[e] [200,200] -> [224,208]
      if (k < 200 && n < 200) v = src[((size_t)e * 200 + k) * 200 + n];
      break;
    case 3:  // W_dist[e] [200,960] -> [224,960]
      if (k < 200) v = src[((size_t)e * 200 + k) * 960 + n];
      break;
    case 4: {  // W_obs [1224,200] -> [1248,208], deter rows 0..199, emb 208..1231
      int kr = (k < 200) ? k : ((k >= HP_ && k < HP_ + 1024) ? (k - 8) : -1);
      if (kr >= 0 && n < 200) v = src[(size_t)kr * 200 + n];
      break; }
    case 5:  // W_od [200,960] -> [224,960]
      if (k < 200) v = src[(size_t)k * 960 + n];
      break;
  }
  dst[(size_t)e * total + p] = f2bf(v);
}

__global__ void init_state(bf16_t* __restrict__ stoch, float* __restrict__ deter,
                           bf16_t* __restrict__ zpad) {
  int i = blockIdx.x * 256 + threadIdx.x;
  if (i < B_ * SD_) stoch[i] = f2bf(0.f);
  if (i < B_ * HP_) deter[i] = 0.f;
  if (i < 64) zpad[i] = f2bf(0.f);
}

// ---------------- fused RSSM step ----------------
// 1 block = 16 batch rows, 4 waves split N-tiles; 64 blocks per step.

__global__ __launch_bounds__(128) void rssm_step(
    int t,
    const float* __restrict__ emb, const float* __restrict__ act,
    const unsigned char* __restrict__ isf,
    const float* __restrict__ b_img, const float* __restrict__ b_gru,
    const float* __restrict__ b_out, const float* __restrict__ b_dist,
    const float* __restrict__ b_obs, const float* __restrict__ b_od,
    const bf16_t* __restrict__ wImg, const bf16_t* __restrict__ wGru,
    const bf16_t* __restrict__ wOut, const bf16_t* __restrict__ wDist,
    const bf16_t* __restrict__ wObs, const bf16_t* __restrict__ wOd,
    const bf16_t* __restrict__ zpad,
    bf16_t* __restrict__ stoch, float* __restrict__ deter,
    float* __restrict__ out) {
  __shared__ __align__(16) bf16_t s_cat[16 * KOBS_];  // [deter(208) | emb(1024) | 0(16)]
  __shared__ __align__(16) bf16_t s_xd[16 * KGRU_];   // [x(208) | deter_prev(208)] -> later ox
  __shared__ __align__(16) bf16_t s_h[16 * KENS_];    // [h(208) | 0(16)]
  __shared__ unsigned char s_fr[16];                  // is_first per block row

  const int tid = threadIdx.x;
  const int lane = tid & 31;
  const int wave = tid >> 5;
  const int row0 = blockIdx.x * 16;

  // ---- per-step PRNG: key(42)=(0,42); fold_in; split; randint ----
  U2 kt = tf2x32(0u, 42u, 0u, (uint32_t)t);
  U2 l0 = tf2x32(kt.x, kt.y, 0u, 2u);
  U2 l1 = tf2x32(kt.x, kt.y, 1u, 3u);
  const uint32_t ki0 = l0.x, ki1 = l1.x;   // k_idx
  const uint32_t kp0 = l0.y, kp1 = l1.y;   // k_post
  const int idx = (int)(tf2x32(ki0, ki1, 0u, 0u).x % (uint32_t)ENS_);

  if (tid < 16) s_fr[tid] = isf[(size_t)(row0 + tid) * T_ + t];
  __syncthreads();

  const int arow = row0 + (lane & 15);
  const bool afirst = s_fr[lane & 15] != 0;
  const float amask = afirst ? 0.f : 1.f;

  // ---- stage 0: stage emb (bf16) + masked deter_prev; zero pads ----
  for (int i = tid; i < 16 * ENC_; i += 128) {
    int r = i >> 10, c = i & 1023;
    s_cat[r * KOBS_ + HP_ + c] = f2bf(emb[((size_t)(row0 + r) * T_ + t) * ENC_ + c]);
  }
  for (int i = tid; i < 16 * 16; i += 128) {
    int r = i >> 4, c = i & 15;
    s_cat[r * KOBS_ + 1232 + c] = f2bf(0.f);
    s_h[r * KENS_ + HP_ + c] = f2bf(0.f);
  }
  for (int i = tid; i < 16 * HP_; i += 128) {
    int r = i / HP_, c = i % HP_;
    float m = s_fr[r] ? 0.f : 1.f;
    s_xd[r * KGRU_ + HP_ + c] = f2bf(m * deter[(size_t)(row0 + r) * HP_ + c]);
  }
  __syncthreads();

  // ---- stage 1: x = elu([stoch*m, act*m] @ W_img + b_img) -> s_xd[:,0:208] ----
  // branch-free masking: masked lanes read a zero page with zero K-step.
  const bf16_t* arb = afirst ? zpad : (stoch + (size_t)arow * SD_);
  const int kstep = afirst ? 0 : 32;
  for (int nt = wave; nt < NT13; nt += 4) {
    v8f acc = zero8();
    for (int kc = 0; kc < 31; ++kc) {
      v16bf a;
      if (kc < 30) {
        a = load_afrag(arb, kc * kstep, lane);
      } else {  // K 960..975 = action * m, 976..991 = zero pad
        int koff = (lane & 16) ? 8 : 0;
#pragma unroll
        for (int j = 0; j < 8; ++j) {
          float av = amask * act[((size_t)arow * T_ + t) * ACT_ + koff + j];
          a[j] = f2bf(av);
          a[8 + j] = f2bf(0.f);
        }
      }
      v16bf b = *(const v16bf*)(wImg + ((size_t)(kc * NT13 + nt) * 32 + lane) * 16);
      acc = wmma_bf16(a, b, acc);
    }
    int col = nt * 16 + (lane & 15);
    float bias = (col < HID_) ? b_img[col] : 0.f;
#pragma unroll
    for (int v = 0; v < 8; ++v) {
      int r = v + ((lane & 16) ? 8 : 0);
      float f = acc[v] + bias;
      f = (f > 0.f) ? f : (__expf(f) - 1.f);
      s_xd[r * KGRU_ + col] = f2bf(f);
    }
  }
  __syncthreads();

  // ---- stage 2: GRU. parts = [x, deter] @ W_gru; deter_new -> ws/out/s_cat ----
  for (int i = wave; i < NT13; i += 4) {
    v8f aR = zero8(), aC = zero8(), aU = zero8();
    const bf16_t* rb = s_xd + (lane & 15) * KGRU_;
    for (int kc = 0; kc < 13; ++kc) {
      v16bf a = load_afrag(rb, kc * 32, lane);
      v16bf bR = *(const v16bf*)(wGru + ((size_t)(kc * NT39 + i) * 32 + lane) * 16);
      v16bf bC = *(const v16bf*)(wGru + ((size_t)(kc * NT39 + 13 + i) * 32 + lane) * 16);
      v16bf bU = *(const v16bf*)(wGru + ((size_t)(kc * NT39 + 26 + i) * 32 + lane) * 16);
      aR = wmma_bf16(a, bR, aR);
      aC = wmma_bf16(a, bC, aC);
      aU = wmma_bf16(a, bU, aU);
    }
    int c = lane & 15;
    int col = i * 16 + c;
    bool real = col < DET_;
    float bR = real ? b_gru[col] : 0.f;
    float bC = real ? b_gru[DET_ + col] : 0.f;
    float bU = real ? b_gru[2 * DET_ + col] : 0.f;
#pragma unroll
    for (int v = 0; v < 8; ++v) {
      int r = v + ((lane & 16) ? 8 : 0);
      int brow = row0 + r;
      float m = s_fr[r] ? 0.f : 1.f;
      float reset = 1.f / (1.f + __expf(-(aR[v] + bR)));
      float cand = tanhf(reset * (aC[v] + bC));
      float upd = 1.f / (1.f + __expf(-(aU[v] + bU - 1.f)));
      float dprev = m * deter[(size_t)brow * HP_ + col];
      float dnew = upd * cand + (1.f - upd) * dprev;
      deter[(size_t)brow * HP_ + col] = dnew;
      s_cat[r * KOBS_ + col] = f2bf(dnew);
      if (real) out[((size_t)brow * T_ + t) * 2080 + 1920 + col] = dnew;
    }
  }
  __syncthreads();

  // ---- stage 3: h = elu(deter @ W_out[idx]+b) -> s_h ; ox = elu(cat @ W_obs+b) -> s_xd ----
  for (int task = wave; task < 26; task += 4) {
    const bf16_t* rb = s_cat + (lane & 15) * KOBS_;
    if (task < NT13) {
      int nt = task;
      v8f acc = zero8();
      const bf16_t* wp = wOut + (size_t)idx * ((size_t)KENS_ * HP_);
      for (int kc = 0; kc < 7; ++kc) {
        v16bf a = load_afrag(rb, kc * 32, lane);
        v16bf b = *(const v16bf*)(wp + ((size_t)(kc * NT13 + nt) * 32 + lane) * 16);
        acc = wmma_bf16(a, b, acc);
      }
      int col = nt * 16 + (lane & 15);
      float bias = (col < HID_) ? b_out[idx * HID_ + col] : 0.f;
#pragma unroll
      for (int v = 0; v < 8; ++v) {
        int r = v + ((lane & 16) ? 8 : 0);
        float f = acc[v] + bias;
        f = (f > 0.f) ? f : (__expf(f) - 1.f);
        s_h[r * KENS_ + col] = f2bf(f);
      }
    } else {
      int nt = task - NT13;
      v8f acc = zero8();
      for (int kc = 0; kc < 39; ++kc) {
        v16bf a = load_afrag(rb, kc * 32, lane);
        v16bf b = *(const v16bf*)(wObs + ((size_t)(kc * NT13 + nt) * 32 + lane) * 16);
        acc = wmma_bf16(a, b, acc);
      }
      int col = nt * 16 + (lane & 15);
      float bias = (col < HID_) ? b_obs[col] : 0.f;
#pragma unroll
      for (int v = 0; v < 8; ++v) {
        int r = v + ((lane & 16) ? 8 : 0);
        float f = acc[v] + bias;
        f = (f > 0.f) ? f : (__expf(f) - 1.f);
        s_xd[r * KGRU_ + col] = f2bf(f);  // ox reuses s_xd[:,0:208]
      }
    }
  }
  __syncthreads();

  // ---- stage 4: prior = h @ W_dist[idx]+b ; post = ox @ W_od+b + categorical ----
  for (int task = wave; task < 90; task += 4) {
    if (task < NT60) {  // prior logits tile
      int nt = task;
      v8f acc = zero8();
      const bf16_t* rb = s_h + (lane & 15) * KENS_;
      const bf16_t* wp = wDist + (size_t)idx * ((size_t)KENS_ * SD_);
      for (int kc = 0; kc < 7; ++kc) {
        v16bf a = load_afrag(rb, kc * 32, lane);
        v16bf b = *(const v16bf*)(wp + ((size_t)(kc * NT60 + nt) * 32 + lane) * 16);
        acc = wmma_bf16(a, b, acc);
      }
      int col = nt * 16 + (lane & 15);
      float bias = b_dist[idx * SD_ + col];
#pragma unroll
      for (int v = 0; v < 8; ++v) {
        int r = v + ((lane & 16) ? 8 : 0);
        out[((size_t)(row0 + r) * T_ + t) * 2080 + SD_ + col] = acc[v] + bias;
      }
    } else {  // posterior group s: tiles 2s, 2s+1 + gumbel argmax over 32
      int s = task - NT60;
      v8f a0 = zero8(), a1 = zero8();
      const bf16_t* rb = s_xd + (lane & 15) * KGRU_;
      for (int kc = 0; kc < 7; ++kc) {
        v16bf a = load_afrag(rb, kc * 32, lane);
        v16bf b0 = *(const v16bf*)(wOd + ((size_t)(kc * NT60 + 2 * s) * 32 + lane) * 16);
        v16bf b1 = *(const v16bf*)(wOd + ((size_t)(kc * NT60 + 2 * s + 1) * 32 + lane) * 16);
        a0 = wmma_bf16(a, b0, a0);
        a1 = wmma_bf16(a, b1, a1);
      }
      int c = lane & 15;
      float bias0 = b_od[32 * s + c];
      float bias1 = b_od[32 * s + 16 + c];
#pragma unroll
      for (int v = 0; v < 8; ++v) {
        int r = v + ((lane & 16) ? 8 : 0);
        int brow = row0 + r;
        float L0 = a0[v] + bias0;
        float L1 = a1[v] + bias1;
        size_t ob = ((size_t)brow * T_ + t) * 2080;
        out[ob + 32 * s + c] = L0;
        out[ob + 32 * s + 16 + c] = L1;
        uint32_t e0 = ((uint32_t)brow * S_ + (uint32_t)s) * D_ + (uint32_t)c;
        float v0 = L0 + gumbel_elem(kp0, kp1, e0);
        float v1 = L1 + gumbel_elem(kp0, kp1, e0 + 16);
        float vm; int dm;
        if (v0 >= v1) { vm = v0; dm = c; } else { vm = v1; dm = 16 + c; }
#pragma unroll
        for (int off = 1; off < 16; off <<= 1) {  // reduce across 16-lane half
          float ov = __shfl_xor(vm, off, 32);
          int od = __shfl_xor(dm, off, 32);
          if (ov > vm || (ov == vm && od < dm)) { vm = ov; dm = od; }
        }
        stoch[(size_t)brow * SD_ + 32 * s + c] = f2bf(c == dm ? 1.f : 0.f);
        stoch[(size_t)brow * SD_ + 32 * s + 16 + c] = f2bf((16 + c) == dm ? 1.f : 0.f);
      }
    }
  }
}

// ---------------- host ----------------

extern "C" void kernel_launch(void* const* d_in, const int* in_sizes, int n_in,
                              void* d_out, int out_size, void* d_ws, size_t ws_size,
                              hipStream_t stream) {
  const float* emb = (const float*)d_in[0];
  const float* act = (const float*)d_in[1];
  const unsigned char* isf = (const unsigned char*)d_in[2];
  const float* W_img = (const float*)d_in[3];
  const float* b_img = (const float*)d_in[4];
  const float* W_gru = (const float*)d_in[5];
  const float* b_gru = (const float*)d_in[6];
  const float* W_out = (const float*)d_in[7];
  const float* b_out = (const float*)d_in[8];
  const float* W_dist = (const float*)d_in[9];
  const float* b_dist = (const float*)d_in[10];
  const float* W_obs = (const float*)d_in[11];
  const float* b_obs = (const float*)d_in[12];
  const float* W_od = (const float*)d_in[13];
  const float* b_od = (const float*)d_in[14];
  float* out = (float*)d_out;
  char* ws = (char*)d_ws;

  constexpr size_t N_WIMG   = (size_t)KIMG_ * HP_;
  constexpr size_t N_WGRU   = (size_t)KGRU_ * NGRU_;
  constexpr size_t N_WOUT1  = (size_t)KENS_ * HP_;
  constexpr size_t N_WDIST1 = (size_t)KENS_ * SD_;
  constexpr size_t N_WOBS   = (size_t)KOBS_ * HP_;
  constexpr size_t N_WOD    = (size_t)KENS_ * SD_;

  constexpr size_t O_WIMG  = 0;
  constexpr size_t O_WGRU  = O_WIMG + N_WIMG * 2;
  constexpr size_t O_WOUT  = O_WGRU + N_WGRU * 2;
  constexpr size_t O_WDIST = O_WOUT + (size_t)ENS_ * N_WOUT1 * 2;
  constexpr size_t O_WOBS  = O_WDIST + (size_t)ENS_ * N_WDIST1 * 2;
  constexpr size_t O_WOD   = O_WOBS + N_WOBS * 2;
  constexpr size_t O_STOCH = O_WOD + N_WOD * 2;
  constexpr size_t O_DETER = O_STOCH + (size_t)B_ * SD_ * 2;
  constexpr size_t O_ZPAD  = O_DETER + (size_t)B_ * HP_ * 4;

  bf16_t* wImg  = (bf16_t*)(ws + O_WIMG);
  bf16_t* wGru  = (bf16_t*)(ws + O_WGRU);
  bf16_t* wOut  = (bf16_t*)(ws + O_WOUT);
  bf16_t* wDist = (bf16_t*)(ws + O_WDIST);
  bf16_t* wObs  = (bf16_t*)(ws + O_WOBS);
  bf16_t* wOd   = (bf16_t*)(ws + O_WOD);
  bf16_t* stoch = (bf16_t*)(ws + O_STOCH);
  float*  deter = (float*)(ws + O_DETER);
  bf16_t* zpad  = (bf16_t*)(ws + O_ZPAD);

  pack_w<<<dim3((unsigned)((N_WIMG + 255) / 256), 1), 256, 0, stream>>>(0, W_img, wImg, (int)N_WIMG);
  pack_w<<<dim3((unsigned)((N_WGRU + 255) / 256), 1), 256, 0, stream>>>(1, W_gru, wGru, (int)N_WGRU);
  pack_w<<<dim3((unsigned)((N_WOUT1 + 255) / 256), ENS_), 256, 0, stream>>>(2, W_out, wOut, (int)N_WOUT1);
  pack_w<<<dim3((unsigned)((N_WDIST1 + 255) / 256), ENS_), 256, 0, stream>>>(3, W_dist, wDist, (int)N_WDIST1);
  pack_w<<<dim3((unsigned)((N_WOBS + 255) / 256), 1), 256, 0, stream>>>(4, W_obs, wObs, (int)N_WOBS);
  pack_w<<<dim3((unsigned)((N_WOD + 255) / 256), 1), 256, 0, stream>>>(5, W_od, wOd, (int)N_WOD);
  init_state<<<(B_ * SD_ + 255) / 256, 256, 0, stream>>>(stoch, deter, zpad);

  for (int t = 0; t < T_; ++t) {
    rssm_step<<<dim3(B_ / 16), 128, 0, stream>>>(
        t, emb, act, isf, b_img, b_gru, b_out, b_dist, b_obs, b_od,
        wImg, wGru, wOut, wDist, wObs, wOd, zpad, stoch, deter, out);
  }
  (void)in_sizes; (void)n_in; (void)out_size; (void)ws_size;
}
